// EmbeddingRowAdapter_1735166788046
// MI455X (gfx1250) — matrix operand order repaired and checked by reading, not compiled, not run on hardware
//
#include <hip/hip_runtime.h>
#include <hip/hip_bf16.h>

typedef __attribute__((ext_vector_type(16))) _Float16 v16h;
typedef __attribute__((ext_vector_type(8)))  float    v8f;

#define D_DIM 256
#define R_DIM 32
#define TOK_PER_BLK 64

// ---------------- pos[] construction (general: does not assume idx==arange) ---
__global__ void init_pos_kernel(int* __restrict__ pos, int V) {
    int i = blockIdx.x * blockDim.x + threadIdx.x;
    if (i < V) pos[i] = -1;
}

__global__ void scatter_pos_kernel(const int* __restrict__ idx,
                                   int* __restrict__ pos, int M, int V) {
    int i = blockIdx.x * blockDim.x + threadIdx.x;
    if (i < M) {
        int v = idx[i];
        if (v >= 0 && v < V) pos[v] = i;
    }
}

// ---------------- WMMA compute + fused gather/store epilogue ------------------
template <bool FULL>
__device__ __forceinline__
void lora_compute(const uint4* __restrict__ sA128,   // 4 uint4 per token row
                  const uint4* __restrict__ sB128,   // 4 uint4 per d column
                  const int*   __restrict__ sId,
                  const float* __restrict__ E,
                  float*       __restrict__ out,
                  int tok0, int T, int tid)
{
    const int lane    = tid & 31;
    const int wave    = tid >> 5;
    const int hi      = (lane >> 4) & 1;   // half-wave selector
    const int mn      = lane & 15;         // A/C row or B/C column in tile
    const int colbase = wave * 32 + mn;    // this lane's column for n-tile 0

    union U16 { v16h v; uint4 q[2]; };

    // B operands for this wave's two adjacent N-tiles (cols colbase, colbase+16).
    // B (32x16) in contiguous k = 16*hi + s order: dwords [col*16 + hi*8 .. +7] -> two b128 loads.
    U16 b0, b1;
    b0.q[0] = sB128[colbase * 4 + hi * 2];
    b0.q[1] = sB128[colbase * 4 + hi * 2 + 1];
    b1.q[0] = sB128[(colbase + 16) * 4 + hi * 2];
    b1.q[1] = sB128[(colbase + 16) * 4 + hi * 2 + 1];

#pragma unroll
    for (int tt = 0; tt < TOK_PER_BLK / 16; ++tt) {
        // A operand, A (16x32) in contiguous k = 16*hi + s order (same as B):
        // dwords {hi*8..hi*8+7} of the token row.
        U16 a;
        const int mrow = tt * 16 + mn;
        a.q[0] = sA128[mrow * 4 + hi * 2];
        a.q[1] = sA128[mrow * 4 + hi * 2 + 1];

        v8f c0 = {}, c1 = {};
        c0 = __builtin_amdgcn_wmma_f32_16x16x32_f16(false, a.v, false, b0.v,
                                                    (short)0, c0, false, false);
        c1 = __builtin_amdgcn_wmma_f32_16x16x32_f16(false, a.v, false, b1.v,
                                                    (short)0, c1, false, false);

        // C/D layout: lane -> N = mn, VGPR i -> M = i + 8*hi.
        const int rbase = tt * 16 + hi * 8;    // first of this lane's 8 token rows

        int idr[8];
#pragma unroll
        for (int i = 0; i < 8; ++i) idr[i] = sId[rbase + i];  // 2x ds_load_b128

        // 32-bit element indices off uniform bases -> SADDR-form vmem.
        const int obase = (tok0 + rbase) * D_DIM + colbase;
#pragma unroll
        for (int i = 0; i < 8; ++i) {
            if (FULL || (tok0 + rbase + i) < T) {
                const int eidx = idr[i] * D_DIM + colbase;
                out[obase + i * D_DIM]      = c0[i] + E[eidx];       // n-tile 0
                out[obase + i * D_DIM + 16] = c1[i] + E[eidx + 16];  // n-tile 1
            }
        }
    }
}

// ---------------- fused embedding gather + LoRA delta via WMMA ----------------
__global__ __launch_bounds__(256)
void lora_embed_kernel(const int*   __restrict__ ids,
                       const int*   __restrict__ pos,
                       const float* __restrict__ E,
                       const float* __restrict__ A_rows,
                       const float* __restrict__ B,
                       float*       __restrict__ out,
                       int T) {
    __shared__ _Float16 sB[D_DIM * R_DIM];        // [d][r] f16 copy of B (16 KB)
    __shared__ _Float16 sA[TOK_PER_BLK * R_DIM];  // [tok][r] selected A rows (4 KB)
    __shared__ int sId[TOK_PER_BLK];
    __shared__ int sP [TOK_PER_BLK];

    const int tid  = threadIdx.x;
    const int tok0 = blockIdx.x * TOK_PER_BLK;

    // Stage B into LDS as f16, layout [d][r] (direct row-major copy).
    for (int i = tid; i < D_DIM * R_DIM; i += 256)
        sB[i] = (_Float16)B[i];

    // ids + adapter-row positions for this block's 64 tokens.
    if (tid < TOK_PER_BLK) {
        int t  = tok0 + tid;
        int id = (t < T) ? ids[t] : 0;
        sId[tid] = id;
        sP [tid] = (t < T) ? pos[id] : -1;
    }
    __syncthreads();

    // Build A_sel tile: 64 tokens x 32 ranks; zero rows for non-adapted tokens.
    for (int i = tid; i < TOK_PER_BLK * R_DIM; i += 256) {
        int tok = i >> 5, r = i & 31;
        int p   = sP[tok];
        sA[i] = (p >= 0) ? (_Float16)A_rows[p * R_DIM + r] : (_Float16)0.0f;
    }
    __syncthreads();

    const uint4* sA128 = (const uint4*)sA;  // 4 uint4 per token row
    const uint4* sB128 = (const uint4*)sB;  // 4 uint4 per d column

    // Block-uniform branch: EXEC stays all-1s inside each path (WMMA-safe).
    if (tok0 + TOK_PER_BLK <= T)
        lora_compute<true >(sA128, sB128, sId, E, out, tok0, T, tid);
    else
        lora_compute<false>(sA128, sB128, sId, E, out, tok0, T, tid);
}

extern "C" void kernel_launch(void* const* d_in, const int* in_sizes, int n_in,
                              void* d_out, int out_size, void* d_ws, size_t ws_size,
                              hipStream_t stream) {
    // setup_inputs() order: ids, idx, E, A_rows, B
    const int*   ids    = (const int*)d_in[0];
    const int*   idx    = (const int*)d_in[1];
    const float* E      = (const float*)d_in[2];
    const float* A_rows = (const float*)d_in[3];
    const float* B      = (const float*)d_in[4];
    float*       out    = (float*)d_out;

    const int T = in_sizes[0];                 // B_SZ * L tokens
    const int M = in_sizes[1];                 // adapted rows
    const int R = in_sizes[3] / M;             // 32
    const int D = in_sizes[4] / R;             // 256
    const int V = in_sizes[2] / D;             // vocab

    int* pos = (int*)d_ws;                     // V ints of scratch

    init_pos_kernel<<<(V + 255) / 256, 256, 0, stream>>>(pos, V);
    scatter_pos_kernel<<<(M + 255) / 256, 256, 0, stream>>>(idx, pos, M, V);

    const int nblk = (T + TOK_PER_BLK - 1) / TOK_PER_BLK;
    lora_embed_kernel<<<nblk, 256, 0, stream>>>(ids, pos, E, A_rows, B, out, T);
}
